// EntropyRecyclingComplexLayer_11338713661877
// MI455X (gfx1250) — matrix operand order repaired and checked by reading, not compiled, not run on hardware
//
#include <hip/hip_runtime.h>
#include <hip/hip_bf16.h>
#include <math.h>

// ---------------------------------------------------------------------------
// EntropyRecyclingComplexLayer for MI455X (gfx1250, wave32, WMMA).
//
//  * Complex GEMMs -> 4 real bf16 WMMA chains, f32 accumulation
//    (v_wmma_f32_16x16x32_bf16), one 16x64 output strip per wave (NT=4).
//  * Weights pre-packed per layer into WMMA B-fragment order (bf16): the hot
//    loop loads each 32x16 fragment as one contiguous 32B v16bf (2x b128,
//    coalesced), zero conversion VALU.  Packed buffer is L2-resident.
//  * Op is store-BW bound on ~400MB broadcast outputs: those are written as
//    16B nontemporal streams; read-once f32 weights use NT loads so the
//    192MB L2 keeps the packed weights + activation planes instead.
// ---------------------------------------------------------------------------

typedef __attribute__((ext_vector_type(16))) __bf16 v16bf;
typedef __attribute__((ext_vector_type(8)))  float  v8f;
typedef __attribute__((ext_vector_type(4)))  float  vf4;
typedef __attribute__((ext_vector_type(2)))  float  vf2;

#define B_   128
#define IN_  1024
#define H_   2048
#define PEN_ 1024
#define P_   16

// ---- A fragment: 16-bit A matrix 16x32 (ISA 7.12.2) -----------------------
static __device__ __forceinline__ v16bf load_a_frag(
    const float* __restrict__ A, int lda, int m0, int k0, int lane) {
  const int half = (lane >> 4) & 1;
  const int row  = lane & 15;
  const float* p = A + (size_t)(m0 + row) * lda + k0;
  v16bf a;
#pragma unroll
  for (int i = 0; i < 8; ++i) a[i]     = (__bf16)p[half * 16 + i];
#pragma unroll
  for (int i = 0; i < 8; ++i) a[8 + i] = (__bf16)p[half * 16 + 8 + i];
  return a;
}

// ---- Weight pre-pack: interleaved complex [K][N] f32 -> two bf16 planes in
// B-fragment order [plane][tn][ks][lane][16]; NT loads (weights read once).
__global__ void pack_weight_kernel(const float* __restrict__ W,
                                   __bf16* __restrict__ out, int K, int N) {
  const int slabs   = K >> 5;
  const int tiles_n = N >> 4;
  const int total   = tiles_n * slabs * 32;
  int tid = blockIdx.x * blockDim.x + threadIdx.x;
  if (tid >= total) return;
  const int lane = tid & 31;
  const int ks   = (tid >> 5) % slabs;
  const int tn   = (tid >> 5) / slabs;
  const int col  = lane & 15;
  const int half = (lane >> 4) & 1;
  const vf2* Wc = (const vf2*)W;
  const size_t base = (size_t)(ks * 32 + half * 16) * N + tn * 16 + col;
  v16bf r, im;
#pragma unroll
  for (int i = 0; i < 16; ++i) {
    vf2 v = __builtin_nontemporal_load(Wc + base + (size_t)i * N);
    r[i]  = (__bf16)v.x;
    im[i] = (__bf16)v.y;
  }
  ((v16bf*)out)[tid] = r;
  ((v16bf*)(out + (size_t)K * N))[tid] = im;
}

// ---- Complex GEMM on packed weights: C[MxN] = A[MxK] * W[KxN] -------------
// Per k-step: A fragments converted once; per tile the WMMA order
//   cr += Ar*Wr ; ci += Ar*Wi ; cr += (-Ai)*Wi ; ci += Ai*Wr
// keeps distance 2 between same-accumulator WMMAs (hazard hiding).
template <int NT>
__global__ void cgemm_pk_wmma_kernel(
    const float* __restrict__ Ar, const float* __restrict__ Ai, int lda,
    const __bf16* __restrict__ Bpk,
    float* __restrict__ Cr, float* __restrict__ Ci,
    int M, int N, int K) {
  const int lane = threadIdx.x & 31;
  const int wave = threadIdx.x >> 5;
  const int wpb  = blockDim.x >> 5;
  const int ngroups = N / (16 * NT);
  const int total   = (M >> 4) * ngroups;
  const int g = blockIdx.x * wpb + wave;
  if (g >= total) return;                         // wave-uniform: EXEC all-1s
  const int m0  = (g / ngroups) << 4;
  const int tn0 = (g % ngroups) * NT;
  const int slabs = K >> 5;
  const v16bf* Br = (const v16bf*)Bpk;
  const v16bf* Bi = (const v16bf*)(Bpk + (size_t)K * N);

  v8f cr[NT] = {};
  v8f ci[NT] = {};
  for (int ks = 0; ks < slabs; ++ks) {
    if (ks + 1 < slabs)  // pull next packed B slab toward the WGP
      __builtin_prefetch(&Br[((size_t)tn0 * slabs + ks + 1) * 32 + lane], 0, 3);
    v16bf ar = load_a_frag(Ar, lda, m0, ks * 32, lane);
    v16bf ai = load_a_frag(Ai, lda, m0, ks * 32, lane);
    v16bf nai;
#pragma unroll
    for (int i = 0; i < 16; ++i) nai[i] = (__bf16)(-(float)ai[i]);
#pragma unroll
    for (int t = 0; t < NT; ++t) {
      const size_t fi = ((size_t)(tn0 + t) * slabs + ks) * 32 + lane;
      v16bf br = Br[fi];
      v16bf bi = Bi[fi];
      cr[t] = __builtin_amdgcn_wmma_f32_16x16x32_bf16(false, ar,  false, br, (short)0, cr[t], false, false);
      ci[t] = __builtin_amdgcn_wmma_f32_16x16x32_bf16(false, ar,  false, bi, (short)0, ci[t], false, false);
      cr[t] = __builtin_amdgcn_wmma_f32_16x16x32_bf16(false, nai, false, bi, (short)0, cr[t], false, false);
      ci[t] = __builtin_amdgcn_wmma_f32_16x16x32_bf16(false, ai,  false, br, (short)0, ci[t], false, false);
    }
  }
  // 32-bit C/D 16x16: VGPR r holds M=r (lanes 0-15) / M=8+r (lanes 16-31)
  const int col   = lane & 15;
  const int rbase = (lane >> 4) ? 8 : 0;
#pragma unroll
  for (int t = 0; t < NT; ++t) {
    const int n0 = (tn0 + t) << 4;
#pragma unroll
    for (int r = 0; r < 8; ++r) {
      Cr[(size_t)(m0 + rbase + r) * N + n0 + col] = cr[t][r];
      Ci[(size_t)(m0 + rbase + r) * N + n0 + col] = ci[t][r];
    }
  }
}

// ---- hidden = hl * sigmoid(|sp - hl|)  (complex * real gate) --------------
__global__ void gate_kernel(const float* __restrict__ hr, const float* __restrict__ hi,
                            const float* __restrict__ sr, const float* __restrict__ si,
                            float* __restrict__ outr, float* __restrict__ outi, int n) {
  int idx = blockIdx.x * blockDim.x + threadIdx.x;
  if (idx >= n) return;
  float a = hr[idx], b = hi[idx];
  float dr = sr[idx] - a, di = si[idx] - b;
  float mag = sqrtf(dr * dr + di * di);
  float g = 1.0f / (1.0f + __expf(-mag));
  outr[idx] = a * g;
  outi[idx] = b * g;
}

// ---- pw = softmax(|attn|) over P=16, one thread per row -------------------
__global__ void absmax_softmax_kernel(const float* __restrict__ ar, const float* __restrict__ ai,
                                      float* __restrict__ pw, int rows) {
  int b = blockIdx.x * blockDim.x + threadIdx.x;
  if (b >= rows) return;
  float m[P_], mx = -1e30f;
#pragma unroll
  for (int p = 0; p < P_; ++p) {
    float x = ar[b * P_ + p], y = ai[b * P_ + p];
    m[p] = sqrtf(x * x + y * y);
    mx = fmaxf(mx, m[p]);
  }
  float s = 0.f;
#pragma unroll
  for (int p = 0; p < P_; ++p) { m[p] = __expf(m[p] - mx); s += m[p]; }
  float inv = 1.0f / s;
#pragma unroll
  for (int p = 0; p < P_; ++p) pw[b * P_ + p] = m[p] * inv;
}

// ---- recon[b,h] = sum_p pw[b,p] * Pd[p,h]   (K=16, plain VALU) ------------
__global__ void recon_kernel(const float* __restrict__ pw, const float* __restrict__ Pd,
                             float* __restrict__ rr, float* __restrict__ ri,
                             int rows, int Hdim) {
  int idx = blockIdx.x * blockDim.x + threadIdx.x;
  if (idx >= rows * Hdim) return;
  int b = idx / Hdim, h = idx % Hdim;
  const vf2* Pc = (const vf2*)Pd;
  float sr = 0.f, si = 0.f;
#pragma unroll
  for (int p = 0; p < P_; ++p) {
    float w = pw[b * P_ + p];
    vf2 d = Pc[(size_t)p * Hdim + h];
    sr += w * d.x; si += w * d.y;
  }
  rr[idx] = sr; ri[idx] = si;
}

// ---- pe[b] = mean_h |(a - c)|^2  (complex diff), block per row ------------
__global__ void pred_error_kernel(const float* __restrict__ ar, const float* __restrict__ ai,
                                  const float* __restrict__ cr, const float* __restrict__ ci,
                                  float* __restrict__ pe, int Hdim) {
  __shared__ float red[256];
  int b = blockIdx.x;
  float s = 0.f;
  for (int h = threadIdx.x; h < Hdim; h += blockDim.x) {
    size_t i = (size_t)b * Hdim + h;
    float dr = ar[i] - cr[i], di = ai[i] - ci[i];
    s += dr * dr + di * di;
  }
  red[threadIdx.x] = s; __syncthreads();
  for (int o = blockDim.x >> 1; o > 0; o >>= 1) {
    if (threadIdx.x < (unsigned)o) red[threadIdx.x] += red[threadIdx.x + o];
    __syncthreads();
  }
  if (threadIdx.x == 0) pe[b] = red[0] / (float)Hdim;
}

// ---- confidence[b] = 0.5*(tanh(-pe*sigmoid(|pe-mean(pe)|))+1) -------------
__global__ void confidence_kernel(const float* __restrict__ pe, float* __restrict__ conf, int n) {
  __shared__ float red[128];
  int t = threadIdx.x;
  red[t] = (t < n) ? pe[t] : 0.f; __syncthreads();
  for (int o = 64; o > 0; o >>= 1) {
    if (t < o) red[t] += red[t + o];
    __syncthreads();
  }
  float mean = red[0] / (float)n;
  if (t < n) {
    float p = pe[t];
    float temp = 1.0f / (1.0f + __expf(-fabsf(p - mean)));
    conf[t] = 0.5f * (tanhf(-p * temp) + 1.0f);
  }
}

// ---- out[i,j,:] = x[j,:] * (1-conf[i] | conf[i]) ; 2 complex/thread,
// 16B nontemporal stream (dominant cost of the whole op) --------------------
__global__ void bcast_scale_kernel(const float* __restrict__ xr, const float* __restrict__ xi,
                                   const float* __restrict__ conf, float* __restrict__ out,
                                   int rows, int inner, int one_minus) {
  size_t idx = (size_t)blockIdx.x * blockDim.x + threadIdx.x;   // complex-pair idx
  size_t per_i2 = (size_t)rows * (inner >> 1);
  size_t total2 = (size_t)rows * per_i2;
  if (idx >= total2) return;
  int i = (int)(idx / per_i2);
  size_t rem2 = idx - (size_t)i * per_i2;       // pair index within [j, inner/2]
  float c = conf[i];
  if (one_minus) c = 1.0f - c;
  vf2 xrp = ((const vf2*)xr)[rem2];
  vf2 xip = ((const vf2*)xi)[rem2];
  vf4 o = {xrp.x * c, xip.x * c, xrp.y * c, xip.y * c};
  __builtin_nontemporal_store(o, (vf4*)out + idx);
}

// ---- entropy = hidden - recon ; 2 complex/thread, 16B NT store ------------
__global__ void entropy_kernel(const float* __restrict__ hr, const float* __restrict__ hi,
                               const float* __restrict__ rr, const float* __restrict__ ri,
                               float* __restrict__ out, int n2) {
  int idx = blockIdx.x * blockDim.x + threadIdx.x;
  if (idx >= n2) return;
  vf2 h0 = ((const vf2*)hr)[idx];
  vf2 h1 = ((const vf2*)hi)[idx];
  vf2 r0 = ((const vf2*)rr)[idx];
  vf2 r1 = ((const vf2*)ri)[idx];
  vf4 o = {h0.x - r0.x, h1.x - r1.x, h0.y - r0.y, h1.y - r1.y};
  __builtin_nontemporal_store(o, (vf4*)out + idx);
}

__global__ void copy_kernel(const float* __restrict__ src, float* __restrict__ dst, int n) {
  int i = blockIdx.x * blockDim.x + threadIdx.x;
  if (i < n) dst[i] = src[i];
}

// ---------------------------------------------------------------------------
extern "C" void kernel_launch(void* const* d_in, const int* in_sizes, int n_in,
                              void* d_out, int out_size, void* d_ws, size_t ws_size,
                              hipStream_t stream) {
  (void)in_sizes; (void)n_in; (void)out_size; (void)ws_size;
  const float* xr  = (const float*)d_in[0];
  const float* xi  = (const float*)d_in[1];
  const float* Wp1 = (const float*)d_in[2];   // complex interleaved [IN,H]
  const float* Ws1 = (const float*)d_in[3];   // [H,H]
  const float* Pd1 = (const float*)d_in[4];   // [P,H]
  const float* Wa1 = (const float*)d_in[5];   // [H,P]
  const float* Wt1 = (const float*)d_in[6];   // [H,PEN]
  const float* Wp2 = (const float*)d_in[7];   // [H,H]
  const float* Ws2 = (const float*)d_in[8];   // [H,H]
  const float* Pd2 = (const float*)d_in[9];   // [P,H]
  const float* Wa2 = (const float*)d_in[10];  // [H,P]
  float* out = (float*)d_out;

  // ---- workspace: f32 planes (~11.6MB) + one reusable packed-weight buffer
  float* w = (float*)d_ws;
  const size_t BH = (size_t)B_ * H_;
  float* hl1r = w + 0 * BH; float* hl1i = w + 1 * BH;   // hl1 -> hidden (in place)
  float* t1r  = w + 2 * BH; float* t1i  = w + 3 * BH;   // sp1 -> hl2 -> actual_next
  float* t2r  = w + 4 * BH; float* t2i  = w + 5 * BH;   // sp2
  float* recr = w + 6 * BH; float* reci = w + 7 * BH;   // recon
  float* r2r  = w + 8 * BH; float* r2i  = w + 9 * BH;   // recon2 (actual_patterns)
  float* pfr  = w + 10 * BH;
  float* pfi  = pfr + (size_t)B_ * PEN_;
  float* attr = pfi + (size_t)B_ * PEN_;
  float* atti = attr + B_ * P_;
  float* pw1  = atti + B_ * P_;
  float* pw2  = pw1 + B_ * P_;
  float* pe   = pw2 + B_ * P_;
  float* conf = pe + B_;
  // packed-weight scratch (bf16, 2 planes, max H*H elems = 16MB), 32B aligned
  const size_t PLANES_FLOATS = 10 * BH + 2 * (size_t)B_ * PEN_ + 4 * (size_t)B_ * P_ + 2 * B_;
  const size_t WPK_OFF = (PLANES_FLOATS + 7) & ~(size_t)7;
  __bf16* wpk = (__bf16*)(w + WPK_OFF);

  const int EB = 256;
  auto pack_grid = [](int K, int N) { return dim3(((N / 16) * (K / 32) * 32 + 255) / 256); };
  auto gemm_grid = [](int M, int N, int NT) { int t = (M / 16) * (N / (16 * NT)); return dim3((t + 3) / 4); };

  // 1) hl1 = x @ Wp1
  pack_weight_kernel<<<pack_grid(IN_, H_), EB, 0, stream>>>(Wp1, wpk, IN_, H_);
  cgemm_pk_wmma_kernel<4><<<gemm_grid(B_, H_, 4), 128, 0, stream>>>(xr, xi, IN_, wpk, hl1r, hl1i, B_, H_, IN_);
  // 2) sp1 = hl1 @ Ws1
  pack_weight_kernel<<<pack_grid(H_, H_), EB, 0, stream>>>(Ws1, wpk, H_, H_);
  cgemm_pk_wmma_kernel<4><<<gemm_grid(B_, H_, 4), 128, 0, stream>>>(hl1r, hl1i, H_, wpk, t1r, t1i, B_, H_, H_);
  // 3) hidden = hl1 * sigmoid(|sp1 - hl1|)   (in place into hl1)
  gate_kernel<<<(int)((BH + EB - 1) / EB), EB, 0, stream>>>(hl1r, hl1i, t1r, t1i, hl1r, hl1i, (int)BH);
  // 4-6) extract layer 1: attn -> softmax -> recon
  pack_weight_kernel<<<pack_grid(H_, P_), EB, 0, stream>>>(Wa1, wpk, H_, P_);
  cgemm_pk_wmma_kernel<1><<<gemm_grid(B_, P_, 1), 128, 0, stream>>>(hl1r, hl1i, H_, wpk, attr, atti, B_, P_, H_);
  absmax_softmax_kernel<<<1, 128, 0, stream>>>(attr, atti, pw1, B_);
  recon_kernel<<<(int)((BH + EB - 1) / EB), EB, 0, stream>>>(pw1, Pd1, recr, reci, B_, H_);
  // 7-9) actual_next = paradox(hidden, Wp2, Ws2)
  pack_weight_kernel<<<pack_grid(H_, H_), EB, 0, stream>>>(Wp2, wpk, H_, H_);
  cgemm_pk_wmma_kernel<4><<<gemm_grid(B_, H_, 4), 128, 0, stream>>>(hl1r, hl1i, H_, wpk, t1r, t1i, B_, H_, H_);
  pack_weight_kernel<<<pack_grid(H_, H_), EB, 0, stream>>>(Ws2, wpk, H_, H_);
  cgemm_pk_wmma_kernel<4><<<gemm_grid(B_, H_, 4), 128, 0, stream>>>(t1r, t1i, H_, wpk, t2r, t2i, B_, H_, H_);
  gate_kernel<<<(int)((BH + EB - 1) / EB), EB, 0, stream>>>(t1r, t1i, t2r, t2i, t1r, t1i, (int)BH);
  // 10-12) extract layer 2 -> actual_patterns (recon2)
  pack_weight_kernel<<<pack_grid(H_, P_), EB, 0, stream>>>(Wa2, wpk, H_, P_);
  cgemm_pk_wmma_kernel<1><<<gemm_grid(B_, P_, 1), 128, 0, stream>>>(t1r, t1i, H_, wpk, attr, atti, B_, P_, H_);
  absmax_softmax_kernel<<<1, 128, 0, stream>>>(attr, atti, pw2, B_);
  recon_kernel<<<(int)((BH + EB - 1) / EB), EB, 0, stream>>>(pw2, Pd2, r2r, r2i, B_, H_);
  // 13-14) pred_error -> confidence
  pred_error_kernel<<<B_, 256, 0, stream>>>(r2r, r2i, recr, reci, pe, H_);
  confidence_kernel<<<1, 128, 0, stream>>>(pe, conf, B_);
  // 15) penult_feat = recon @ Wt1
  pack_weight_kernel<<<pack_grid(H_, PEN_), EB, 0, stream>>>(Wt1, wpk, H_, PEN_);
  cgemm_pk_wmma_kernel<4><<<gemm_grid(B_, PEN_, 4), 128, 0, stream>>>(recr, reci, H_, wpk, pfr, pfi, B_, PEN_, H_);

  // 16) outputs (flat, return order; complex64 as interleaved float pairs)
  size_t n_cu = (size_t)B_ * B_ * H_;                 // continue_up [B,B,1,H]
  size_t n_pn = (size_t)B_ * B_ * PEN_;               // penult      [B,B,1,PEN]
  size_t o1 = 2 * n_cu, o2 = o1 + 2 * n_pn, o3 = o2 + B_;
  size_t cu2 = n_cu / 2, pn2 = n_pn / 2;              // 16B (2-complex) units
  bcast_scale_kernel<<<(int)((cu2 + EB - 1) / EB), EB, 0, stream>>>(hl1r, hl1i, conf, out, B_, H_, 1);
  bcast_scale_kernel<<<(int)((pn2 + EB - 1) / EB), EB, 0, stream>>>(pfr, pfi, conf, out + o1, B_, PEN_, 0);
  copy_kernel<<<1, 128, 0, stream>>>(pe, out + o2, B_);
  entropy_kernel<<<(int)((BH / 2 + EB - 1) / EB), EB, 0, stream>>>(hl1r, hl1i, recr, reci, out + o3, (int)(BH / 2));
}
